// META1_74569222193913
// MI455X (gfx1250) — matrix operand order repaired and checked by reading, not compiled, not run on hardware
//
#include <hip/hip_runtime.h>

typedef __attribute__((ext_vector_type(16))) _Float16 v16h;
typedef __attribute__((ext_vector_type(8)))  _Float16 v8h;
typedef __attribute__((ext_vector_type(8)))  float    v8f;
typedef __attribute__((ext_vector_type(4)))  float    v4f;

#define D2c 50
#define D3c 15
#define TPW 4   // tiles (16 edges) per wave

__device__ __forceinline__ v8f wmma_f16(v16h a, v16h b, v8f c) {
  return __builtin_amdgcn_wmma_f32_16x16x32_f16(false, a, false, b, (short)0, c, false, false);
}

__global__ void zero_kernel(float* p, int n) {
  int i = blockIdx.x * blockDim.x + threadIdx.x;
  if (i < n) p[i] = 0.0f;
}

// Fused edge-MLP + node-MLP + scatter, computed transposed: Ht = Wt x Xt.
// A-fragments = pre-swizzled weights (contiguous per lane).
// B-fragments = row-major activations (contiguous per lane).
// C results   = lane-owned edge columns (contiguous b128 stores).
__global__ __launch_bounds__(256) void edge_node_kernel(
    const float* __restrict__ x, const int* __restrict__ eidx,
    const float* __restrict__ eattr,
    const float* __restrict__ ew1, const float* __restrict__ eb1,
    const float* __restrict__ ew2, const float* __restrict__ eb2,
    const float* __restrict__ nw1, const float* __restrict__ nb1,
    const float* __restrict__ nw2, const float* __restrict__ nb2,
    float* __restrict__ x2sum, float* __restrict__ cnt, int E)
{
  // 12 weight A-fragments: [f][lane][h] contiguous 16 halves per lane
  // f0-3: W1e^T tiles | f4-5: W2e^T K-halves | f6-9: W1n^T | f10-11: W2n^T
  __shared__ alignas(32) _Float16 WF[12 * 32 * 16];
  __shared__ alignas(16) float eb1p[64], nb1p[64], eb2p[16], nb2p[16];
  __shared__ alignas(32) _Float16 hid[8][16 * 64];   // per-wave H (16 edges x 64)
  __shared__ alignas(32) _Float16 e2s[8][16 * 16];   // per-wave edge2 (16 x 16)
  __shared__ alignas(32) _Float16 feat[8][16 * 32];  // per-wave inputs (16 x 32)
  __shared__ int rowL[8][16];

  int tid = threadIdx.x;

  // ---- stage pre-swizzled weight A-fragments (this code's contiguous k order: lane<16 h0-15->K0-15 ;
  // lane>=16 -> K16-31 ; M row = lane&15)
  for (int i = tid; i < 12 * 32 * 16; i += 256) {
    int f = i >> 9;
    int ln = (i >> 4) & 31;
    int h = i & 15;
    int k = ((ln < 16) ? 0 : 16) + h;
    int p = ln & 15;
    float v = 0.0f;
    if (f < 4)       { int o = f * 16 + p;         if (k < 9  && o < D2c) v = ew1[k * D2c + o]; }
    else if (f < 6)  { int kk = (f - 4) * 32 + k;  if (kk < D2c && p < D3c) v = ew2[kk * D3c + p]; }
    else if (f < 10) { int o = (f - 6) * 16 + p;   if (k < 18 && o < D2c) v = nw1[k * D2c + o]; }
    else             { int kk = (f - 10) * 32 + k; if (kk < D2c && p < D3c) v = nw2[kk * D3c + p]; }
    WF[i] = (_Float16)v;
  }
  if (tid < 64) {
    eb1p[tid] = (tid < D2c) ? eb1[tid] : 0.0f;
    nb1p[tid] = (tid < D2c) ? nb1[tid] : 0.0f;
  }
  if (tid < 16) {
    eb2p[tid] = (tid < D3c) ? eb2[tid] : 0.0f;
    nb2p[tid] = (tid < D3c) ? nb2[tid] : 0.0f;
  }
  __syncthreads();

  int wv = tid >> 5;
  int lane = tid & 31;
  int m16 = lane & 15;
  bool lo = (lane < 16);
  int hi8 = lo ? 0 : 8;               // C row offset / feature-group offset
  int kb16 = lo ? 0 : 16;             // B-fragment K half offset
  const v16h* WFv = (const v16h*)WF;  // WFv[f*32 + lane]

  for (int it = 0; it < TPW; ++it) {
    int base = ((blockIdx.x * 8 + wv) * TPW + it) * 16;

    // ---- stage edge inputs [x[row] | x[col] | edge_attr] into feat (16x32)
    float xc0 = 0.0f, xc1 = 0.0f, xc2 = 0.0f;
    if (lo) {
      int e = base + lane;
      bool vld = (e < E);
      int ec = vld ? e : 0;
      int r = eidx[ec];
      int c = eidx[E + ec];
      rowL[wv][lane] = vld ? r : -1;
      float f0=0,f1=0,f2=0,f3=0,f4=0,f5=0,f6=0,f7=0,f8=0;
      if (vld) {
        f0 = x[r * 3 + 0]; f1 = x[r * 3 + 1]; f2 = x[r * 3 + 2];
        f3 = x[c * 3 + 0]; f4 = x[c * 3 + 1]; f5 = x[c * 3 + 2];
        f6 = eattr[e * 3 + 0]; f7 = eattr[e * 3 + 1]; f8 = eattr[e * 3 + 2];
      }
      xc0 = f3; xc1 = f4; xc2 = f5;
      v8h q0 = {(_Float16)f0,(_Float16)f1,(_Float16)f2,(_Float16)f3,
                (_Float16)f4,(_Float16)f5,(_Float16)f6,(_Float16)f7};
      v8h q1 = {};
      q1[0] = (_Float16)f8;
      v8h zz = {};
      v8h* fp = (v8h*)&feat[wv][lane * 32];
      fp[0] = q0; fp[1] = q1; fp[2] = zz; fp[3] = zz;
    }

    // ---- edge layer 1: Ht = W1e^T (4 M-tiles) x Xt   -> hid (relu+bias)
    {
      v16h bx = *(const v16h*)&feat[wv][m16 * 32 + kb16];
#pragma unroll
      for (int t = 0; t < 4; ++t) {
        v16h aw = WFv[t * 32 + lane];
        v8f c = {};
        c = wmma_f16(aw, bx, c);
        const v4f* bp = (const v4f*)&eb1p[t * 16 + hi8];
        v4f b0 = bp[0], b1 = bp[1];
        v8h hv;
#pragma unroll
        for (int r = 0; r < 8; ++r) {
          float bias = (r < 4) ? b0[r] : b1[r - 4];
          hv[r] = (_Float16)fmaxf(c[r] + bias, 0.0f);
        }
        *(v8h*)&hid[wv][m16 * 64 + t * 16 + hi8] = hv;
      }
    }

    // ---- edge layer 2: edge2^T = W2e^T x Ht   -> e2s
    {
      v16h b0f = *(const v16h*)&hid[wv][m16 * 64 + kb16];
      v16h b1f = *(const v16h*)&hid[wv][m16 * 64 + 32 + kb16];
      v8f c = {};
      c = wmma_f16(WFv[4 * 32 + lane], b0f, c);
      c = wmma_f16(WFv[5 * 32 + lane], b1f, c);
      const v4f* bp = (const v4f*)&eb2p[hi8];
      v4f b0 = bp[0], b1 = bp[1];
      v8h ev;
#pragma unroll
      for (int r = 0; r < 8; ++r) {
        float bias = (r < 4) ? b0[r] : b1[r - 4];
        ev[r] = (_Float16)(c[r] + bias);   // cols >= 15 are exactly 0 (zero pad)
      }
      *(v8h*)&e2s[wv][m16 * 16 + hi8] = ev;
    }

    // ---- restage node input [x[col] | edge2] into feat
    if (lo) {
      const v8h* ep = (const v8h*)&e2s[wv][lane * 16];
      v8h e0 = ep[0], e1 = ep[1];
      v8h q0, q1, q2 = {}, q3 = {};
      q0[0] = (_Float16)xc0; q0[1] = (_Float16)xc1; q0[2] = (_Float16)xc2;
      q0[3] = e0[0]; q0[4] = e0[1]; q0[5] = e0[2]; q0[6] = e0[3]; q0[7] = e0[4];
      q1[0] = e0[5]; q1[1] = e0[6]; q1[2] = e0[7];
      q1[3] = e1[0]; q1[4] = e1[1]; q1[5] = e1[2]; q1[6] = e1[3]; q1[7] = e1[4];
      q2[0] = e1[5]; q2[1] = e1[6];
      v8h* fp = (v8h*)&feat[wv][lane * 32];
      fp[0] = q0; fp[1] = q1; fp[2] = q2; fp[3] = q3;
    }

    // ---- node layer 1
    {
      v16h bx = *(const v16h*)&feat[wv][m16 * 32 + kb16];
#pragma unroll
      for (int t = 0; t < 4; ++t) {
        v16h aw = WFv[(6 + t) * 32 + lane];
        v8f c = {};
        c = wmma_f16(aw, bx, c);
        const v4f* bp = (const v4f*)&nb1p[t * 16 + hi8];
        v4f b0 = bp[0], b1 = bp[1];
        v8h hv;
#pragma unroll
        for (int r = 0; r < 8; ++r) {
          float bias = (r < 4) ? b0[r] : b1[r - 4];
          hv[r] = (_Float16)fmaxf(c[r] + bias, 0.0f);
        }
        *(v8h*)&hid[wv][m16 * 64 + t * 16 + hi8] = hv;
      }
    }

    // ---- node layer 2 -> msg^T ; lane owns edge m16, features hi8..hi8+7
    {
      v16h b0f = *(const v16h*)&hid[wv][m16 * 64 + kb16];
      v16h b1f = *(const v16h*)&hid[wv][m16 * 64 + 32 + kb16];
      v8f c = {};
      c = wmma_f16(WFv[10 * 32 + lane], b0f, c);
      c = wmma_f16(WFv[11 * 32 + lane], b1f, c);
      int row = rowL[wv][m16];
      const v4f* bp = (const v4f*)&nb2p[hi8];
      v4f b0 = bp[0], b1 = bp[1];
      if (row >= 0) {
#pragma unroll
        for (int r = 0; r < 8; ++r) {
          float val = c[r] + ((r < 4) ? b0[r] : b1[r - 4]);
          if (hi8 + r < D3c)   // compile-time true for r<7; skips j==15
            atomicAdd(&x2sum[row * D3c + hi8 + r], val);
        }
        if (lo) atomicAdd(&cnt[row], 1.0f);
      }
    }
  }
}

// per-node mean then scatter into per-graph sums
__global__ void node_kernel(const float* __restrict__ x2sum, const float* __restrict__ cnt,
                            const int* __restrict__ batch,
                            float* __restrict__ gsum, float* __restrict__ gcnt, int N) {
  int n = blockIdx.x * blockDim.x + threadIdx.x;
  if (n >= N) return;
  float inv = 1.0f / fmaxf(cnt[n], 1.0f);
  int g = batch[n];
#pragma unroll
  for (int j = 0; j < 15; ++j)
    atomicAdd(&gsum[g * 15 + j], x2sum[n * 15 + j] * inv);
  atomicAdd(&gcnt[g], 1.0f);
}

// global MLP + fc1 + batch-stat BN + ReLU + fc2 + log_softmax (single block)
__global__ __launch_bounds__(256) void head_kernel(
    const float* __restrict__ u, const float* __restrict__ gsum, const float* __restrict__ gcnt,
    const float* __restrict__ gw1, const float* __restrict__ gb1,
    const float* __restrict__ gw2, const float* __restrict__ gb2,
    const float* __restrict__ f1w, const float* __restrict__ f1b,
    const float* __restrict__ bng, const float* __restrict__ bnb,
    const float* __restrict__ f2w, const float* __restrict__ f2b,
    float* __restrict__ out, int G)
{
  __shared__ float z[1024 * 10];
  __shared__ float ssum[10], ssq[10];
  int tid = threadIdx.x;
  if (tid < 10) { ssum[tid] = 0.0f; ssq[tid] = 0.0f; }
  __syncthreads();
  int Gc = (G < 1024) ? G : 1024;
  float ls[10], lq[10];
#pragma unroll
  for (int o = 0; o < 10; ++o) { ls[o] = 0.0f; lq[o] = 0.0f; }
  for (int g = tid; g < Gc; g += 256) {
    float gi[16];
    gi[0] = u[g];
    float inv = 1.0f / fmaxf(gcnt[g], 1.0f);
    for (int j = 0; j < 15; ++j) gi[1 + j] = gsum[g * 15 + j] * inv;
    float h[50];
    for (int o = 0; o < 50; ++o) {
      float s = gb1[o];
      for (int k = 0; k < 16; ++k) s += gi[k] * gw1[k * 50 + o];
      h[o] = fmaxf(s, 0.0f);
    }
    float u2[15];
    for (int o = 0; o < 15; ++o) {
      float s = gb2[o];
      for (int k = 0; k < 50; ++k) s += h[k] * gw2[k * 15 + o];
      u2[o] = s;
    }
    for (int o = 0; o < 10; ++o) {
      float s = f1b[o];
      for (int k = 0; k < 15; ++k) s += u2[k] * f1w[k * 10 + o];
      z[g * 10 + o] = s;
      ls[o] += s; lq[o] += s * s;
    }
  }
#pragma unroll
  for (int o = 0; o < 10; ++o) { atomicAdd(&ssum[o], ls[o]); atomicAdd(&ssq[o], lq[o]); }
  __syncthreads();
  float invG = 1.0f / (float)((Gc > 0) ? Gc : 1);
  for (int g = tid; g < Gc; g += 256) {
    float zn[10];
    for (int o = 0; o < 10; ++o) {
      float mean = ssum[o] * invG;
      float var  = ssq[o] * invG - mean * mean;   // biased, matches torch BN
      float v = (z[g * 10 + o] - mean) * rsqrtf(var + 1e-5f) * bng[o] + bnb[o];
      zn[o] = fmaxf(v, 0.0f);
    }
    float lg[6]; float mx = -1e30f;
    for (int c = 0; c < 6; ++c) {
      float s = f2b[c];
      for (int k = 0; k < 10; ++k) s += zn[k] * f2w[k * 6 + c];
      lg[c] = s; mx = fmaxf(mx, s);
    }
    float se = 0.0f;
    for (int c = 0; c < 6; ++c) se += expf(lg[c] - mx);
    float lse = mx + logf(se);
    for (int c = 0; c < 6; ++c) out[g * 6 + c] = lg[c] - lse;
  }
}

extern "C" void kernel_launch(void* const* d_in, const int* in_sizes, int n_in,
                              void* d_out, int out_size, void* d_ws, size_t ws_size,
                              hipStream_t stream) {
  const float* x     = (const float*)d_in[0];
  const int*   eidx  = (const int*)d_in[1];
  const float* eattr = (const float*)d_in[2];
  const float* u     = (const float*)d_in[3];
  const int*   batch = (const int*)d_in[4];
  const float* ew1 = (const float*)d_in[5];
  const float* eb1 = (const float*)d_in[6];
  const float* ew2 = (const float*)d_in[7];
  const float* eb2 = (const float*)d_in[8];
  const float* nw1 = (const float*)d_in[9];
  const float* nb1 = (const float*)d_in[10];
  const float* nw2 = (const float*)d_in[11];
  const float* nb2 = (const float*)d_in[12];
  const float* gw1 = (const float*)d_in[13];
  const float* gb1 = (const float*)d_in[14];
  const float* gw2 = (const float*)d_in[15];
  const float* gb2 = (const float*)d_in[16];
  const float* f1w = (const float*)d_in[17];
  const float* f1b = (const float*)d_in[18];
  const float* bng = (const float*)d_in[19];
  const float* bnb = (const float*)d_in[20];
  const float* f2w = (const float*)d_in[21];
  const float* f2b = (const float*)d_in[22];

  int N = in_sizes[0] / 3;
  int E = in_sizes[1] / 2;
  int G = in_sizes[3];

  float* ws    = (float*)d_ws;
  float* x2sum = ws;                          // N*15
  float* cnt   = x2sum + (size_t)N * 15;      // N
  float* gsum  = cnt + N;                     // G*15
  float* gcnt  = gsum + (size_t)G * 15;       // G

  int ztot = N * 16 + G * 16;
  zero_kernel<<<(ztot + 255) / 256, 256, 0, stream>>>(ws, ztot);

  int tiles  = (E + 15) / 16;
  int blocks = (tiles + 8 * TPW - 1) / (8 * TPW);
  edge_node_kernel<<<blocks, 256, 0, stream>>>(x, eidx, eattr,
      ew1, eb1, ew2, eb2, nw1, nb1, nw2, nb2, x2sum, cnt, E);

  node_kernel<<<(N + 255) / 256, 256, 0, stream>>>(x2sum, cnt, batch, gsum, gcnt, N);

  head_kernel<<<1, 256, 0, stream>>>(u, gsum, gcnt, gw1, gb1, gw2, gb2,
      f1w, f1b, bng, bnb, f2w, f2b, (float*)d_out, G);
}
